// EdgeGraphSAGE_44444321579080
// MI455X (gfx1250) — compile-verified, matrix operand order checked by reading, not run on hardware
//
#include <hip/hip_runtime.h>
#include <hip/hip_bf16.h>

#define NODES 50000
#define DEG   16
#define EDGES (NODES * DEG)
#define CH    128
#define GATES 512
#define EDIM  16
#define K1REAL 272   // 2*H + ED
#define K1PAD  288   // padded to 9 K-tiles of 32

typedef __attribute__((ext_vector_type(16))) _Float16 v16h;
typedef __attribute__((ext_vector_type(8)))  _Float16 v8h;
typedef __attribute__((ext_vector_type(8)))  float    v8f;
typedef __attribute__((ext_vector_type(4)))  float    v4f;

__device__ __forceinline__ float sigf(float x) { return 1.0f / (1.0f + __expf(-x)); }

// ---- CDNA5 async memory->LDS copy (16B per lane), ASYNCcnt tracked ----------
__device__ __forceinline__ void async_copy_b128(void* lds_dst, const void* gsrc) {
  asm volatile("global_load_async_to_lds_b128 %0, %1, off"
               :: "v"((unsigned)(uintptr_t)lds_dst),
                  "v"((unsigned long long)(uintptr_t)gsrc)
               : "memory");
}
__device__ __forceinline__ void wait_async() {
  asm volatile("s_wait_asynccnt 0" ::: "memory");
}

// ---- WMMA helpers -----------------------------------------------------------
__device__ __forceinline__ v8f wmma_f16(v16h a, v16h b, v8f c) {
  return __builtin_amdgcn_wmma_f32_16x16x32_f16(false, a, false, b, (short)0, c,
                                                false, false);
}

// A-fragment from a row-major f16 LDS tile [16][ldk] (ISA 7.12.2 lane striping).
__device__ __forceinline__ v16h load_a_frag(const _Float16* s, int m, int kt,
                                            int hi_half, int ldk) {
  int kb = kt * 32 + hi_half * 8;
  v8h lo = *(const v8h*)(s + m * ldk + kb);
  v8h hi = *(const v8h*)(s + m * ldk + kb + 16);
  v16h r;
#pragma unroll
  for (int i = 0; i < 8; ++i) { r[i] = lo[i]; r[i + 8] = hi[i]; }
  return r;
}

// B-fragment from pre-packed fragment-major global buffer (1024 halves/frag).
__device__ __forceinline__ v16h load_b_frag(const _Float16* __restrict__ p,
                                            int frag, int lane) {
  return *(const v16h*)(p + (size_t)frag * 1024 + lane * 16);
}

// pack 8 floats -> v8h
__device__ __forceinline__ v8h cvt8(v4f a, v4f b) {
  v8h r;
#pragma unroll
  for (int i = 0; i < 4; ++i) { r[i] = (_Float16)a[i]; r[i + 4] = (_Float16)b[i]; }
  return r;
}

// ---- Weight packing: W[Nout][Kreal] row-major -> f16 B-fragment layout ------
__global__ void pack_b_frag(const float* __restrict__ W, _Float16* __restrict__ out,
                            int Kreal, int Kpad, int Nout) {
  int idx = blockIdx.x * 256 + threadIdx.x;
  int total = Kpad * Nout;
  if (idx >= total) return;
  int frag   = idx >> 10;
  int within = idx & 1023;
  int lane = within >> 4;
  int e    = within & 15;
  int nkt = Kpad >> 5;
  int nt = frag / nkt;
  int kt = frag - nt * nkt;
  int n  = nt * 16 + (lane & 15);
  int kk = (e < 8 ? e : e + 8) + ((lane >> 4) << 3);
  int k  = kt * 32 + kk;
  float v = (k < Kreal) ? W[(size_t)n * Kreal + k] : 0.0f;
  out[idx] = (_Float16)v;
}

// ---- One SAGEConv(LSTM-aggr) layer: 16 nodes per workgroup, 8 waves ---------
__global__ void __launch_bounds__(256, 1)
sage_lstm_layer(const float* __restrict__ in_f32, const _Float16* __restrict__ in_f16,
                int in_is_f32, const int* __restrict__ src,
                const _Float16* __restrict__ wih_p, const _Float16* __restrict__ whh_p,
                const float* __restrict__ bih, const float* __restrict__ bhh,
                const _Float16* __restrict__ linl_p, const float* __restrict__ linl_b,
                const _Float16* __restrict__ linr_p,
                _Float16* __restrict__ h_out) {
  __shared__ __align__(16) _Float16 s_x[16 * CH];    // self features (lin_r A)
  __shared__ __align__(16) _Float16 s_xt[16 * CH];   // current neighbor tile
  __shared__ __align__(16) _Float16 s_h[16 * CH];    // LSTM hidden (f16 A)
  __shared__ __align__(16) float s_c[16 * CH];       // LSTM cell state
  __shared__ __align__(16) float s_g[16 * GATES];    // gate staging

  const int tid  = threadIdx.x;
  const int lane = tid & 31;
  const int w    = tid >> 5;
  const int node0 = blockIdx.x * 16;
  const int row = tid >> 4;        // staging: 16 threads per row
  const int c0  = (tid & 15) * 8;  // each thread moves 8 halves (16B)

  // stage self features, zero h/c (vectorized)
  {
    int n = node0 + row;
    if (in_is_f32) {
      const float* gp = in_f32 + (size_t)n * CH + c0;
      *(v8h*)(s_x + row * CH + c0) = cvt8(*(const v4f*)gp, *(const v4f*)(gp + 4));
    } else {
      async_copy_b128(s_x + row * CH + c0, in_f16 + (size_t)n * CH + c0);
    }
    v8h z = {};
    *(v8h*)(s_h + row * CH + c0) = z;
    v4f zf = {};
    *(v4f*)(s_c + row * CH + c0) = zf;
    *(v4f*)(s_c + row * CH + c0 + 4) = zf;
  }

  // Register-resident B fragments: wave owns gate columns [w*64, w*64+64).
  v16h wihf[4][4], whhf[4][4];
#pragma unroll
  for (int nt = 0; nt < 4; ++nt)
#pragma unroll
    for (int kt = 0; kt < 4; ++kt) {
      int fr = (w * 4 + nt) * 4 + kt;
      wihf[nt][kt] = load_b_frag(wih_p, fr, lane);
      whhf[nt][kt] = load_b_frag(whh_p, fr, lane);
    }

  const int m  = lane & 15;
  const int hh = lane >> 4;

  // per-column gate bias, loop-invariant over the 16 steps
  float bsum[4];
#pragma unroll
  for (int nt = 0; nt < 4; ++nt) {
    int col = w * 64 + nt * 16 + m;
    bsum[nt] = bih[col] + bhh[col];
  }

  for (int t = 0; t < DEG; ++t) {
    // gather t-th neighbor row for each of the 16 nodes (edges sorted by dst)
    {
      int eid = (node0 + row) * DEG + t;
      int sn  = src[eid];
      if (in_is_f32) {
        const float* gp = in_f32 + (size_t)sn * CH + c0;
        *(v8h*)(s_xt + row * CH + c0) = cvt8(*(const v4f*)gp, *(const v4f*)(gp + 4));
      } else {
        async_copy_b128(s_xt + row * CH + c0, in_f16 + (size_t)sn * CH + c0);
      }
    }
    wait_async();
    __syncthreads();

    // gates = xt @ Wih^T + h @ Whh^T + bih + bhh
    v16h ax[4], ah[4];
#pragma unroll
    for (int kt = 0; kt < 4; ++kt) {
      ax[kt] = load_a_frag(s_xt, m, kt, hh, CH);
      ah[kt] = load_a_frag(s_h,  m, kt, hh, CH);
    }
#pragma unroll
    for (int nt = 0; nt < 4; ++nt) {
      v8f acc = {};
#pragma unroll
      for (int kt = 0; kt < 4; ++kt) {
        acc = wmma_f16(ax[kt], wihf[nt][kt], acc);
        acc = wmma_f16(ah[kt], whhf[nt][kt], acc);
      }
      int col = w * 64 + nt * 16 + m;
#pragma unroll
      for (int r = 0; r < 8; ++r) {
        int mm = r + hh * 8;   // C/D layout: VGPR r -> row r (lanes<16) / r+8
        s_g[mm * GATES + col] = acc[r] + bsum[nt];
      }
    }
    __syncthreads();

    // fused LSTM cell update (torch gate order i,f,g,o), vectorized LDS access
    {
      int base = tid * 8;
      int mm = base >> 7;
      int cc0 = base & 127;
      const float* gp = s_g + mm * GATES + cc0;
      v4f gi0 = *(const v4f*)(gp);           v4f gi1 = *(const v4f*)(gp + 4);
      v4f gf0 = *(const v4f*)(gp + 128);     v4f gf1 = *(const v4f*)(gp + 132);
      v4f gg0 = *(const v4f*)(gp + 256);     v4f gg1 = *(const v4f*)(gp + 260);
      v4f go0 = *(const v4f*)(gp + 384);     v4f go1 = *(const v4f*)(gp + 388);
      v4f cp0 = *(const v4f*)(s_c + mm * CH + cc0);
      v4f cp1 = *(const v4f*)(s_c + mm * CH + cc0 + 4);
      v4f cn0, cn1;
      v8h hv;
#pragma unroll
      for (int j = 0; j < 4; ++j) {
        float cA = sigf(gf0[j]) * cp0[j] + sigf(gi0[j]) * tanhf(gg0[j]);
        float cB = sigf(gf1[j]) * cp1[j] + sigf(gi1[j]) * tanhf(gg1[j]);
        cn0[j] = cA;
        cn1[j] = cB;
        hv[j]     = (_Float16)(sigf(go0[j]) * tanhf(cA));
        hv[j + 4] = (_Float16)(sigf(go1[j]) * tanhf(cB));
      }
      *(v4f*)(s_c + mm * CH + cc0) = cn0;
      *(v4f*)(s_c + mm * CH + cc0 + 4) = cn1;
      *(v8h*)(s_h + mm * CH + cc0) = hv;
    }
    __syncthreads();
  }

  // out = relu(agg @ lin_l^T + b_l + x @ lin_r^T); wave w -> cols [w*16, w*16+16)
  {
    v8f acc = {};
#pragma unroll
    for (int kt = 0; kt < 4; ++kt) {
      v16h ah2 = load_a_frag(s_h, m, kt, hh, CH);
      v16h ax2 = load_a_frag(s_x, m, kt, hh, CH);
      acc = wmma_f16(ah2, load_b_frag(linl_p, w * 4 + kt, lane), acc);
      acc = wmma_f16(ax2, load_b_frag(linr_p, w * 4 + kt, lane), acc);
    }
    int col = w * 16 + m;
    float b = linl_b[col];
#pragma unroll
    for (int r = 0; r < 8; ++r) {
      int mm = r + hh * 8;
      float v = acc[r] + b;
      v = v > 0.f ? v : 0.f;
      h_out[(size_t)(node0 + mm) * CH + col] = (_Float16)v;
    }
  }
}

// ---- Edge MLP: 16 edges per workgroup, gather-concat + 2 WMMA GEMMs ---------
__global__ void __launch_bounds__(256, 1)
edge_mlp(const _Float16* __restrict__ h,
         const int* __restrict__ src, const int* __restrict__ dst,
         const float* __restrict__ eattr,
         const _Float16* __restrict__ w1_p, const float* __restrict__ b1,
         const _Float16* __restrict__ w2_p, const float* __restrict__ b2,
         const float* __restrict__ W3, const float* __restrict__ b3,
         float* __restrict__ out) {
  __shared__ __align__(16) _Float16 s_f[16 * K1PAD];  // [h_src | h_dst | attr | pad]
  __shared__ __align__(16) _Float16 s_h1[16 * CH];
  __shared__ __align__(16) float s_h2[16 * 64];
  __shared__ __align__(16) float s_w3[2 * 64];

  const int tid  = threadIdx.x;
  const int lane = tid & 31;
  const int w    = tid >> 5;
  const int e0   = blockIdx.x * 16;
  const int row  = tid >> 4;
  const int t16  = tid & 15;

  {
    int eid = e0 + row;
    int sn = src[eid];
    int dn = dst[eid];
    // async direct-to-LDS copies of the two f16 feature rows (16B per lane)
    async_copy_b128(s_f + row * K1PAD + t16 * 8,      h + (size_t)sn * CH + t16 * 8);
    async_copy_b128(s_f + row * K1PAD + CH + t16 * 8, h + (size_t)dn * CH + t16 * 8);
    if (t16 < 2) {          // edge_attr: 16 floats -> f16
      const float* ap = eattr + (size_t)eid * EDIM + t16 * 8;
      *(v8h*)(s_f + row * K1PAD + 2 * CH + t16 * 8) =
          cvt8(*(const v4f*)ap, *(const v4f*)(ap + 4));
    } else if (t16 < 4) {   // zero pad 272..287
      v8h z = {};
      *(v8h*)(s_f + row * K1PAD + K1REAL + (t16 - 2) * 8) = z;
    }
    if (tid < 128) s_w3[tid] = W3[tid];
  }
  wait_async();
  __syncthreads();

  const int m  = lane & 15;
  const int hh = lane >> 4;

  // layer 1: [16,288] x [288,128], wave w -> cols [w*16, w*16+16)
  {
    v8f acc = {};
#pragma unroll
    for (int kt = 0; kt < 9; ++kt) {
      v16h a = load_a_frag(s_f, m, kt, hh, K1PAD);
      acc = wmma_f16(a, load_b_frag(w1_p, w * 9 + kt, lane), acc);
    }
    int col = w * 16 + m;
    float b = b1[col];
#pragma unroll
    for (int r = 0; r < 8; ++r) {
      int mm = r + hh * 8;
      float v = acc[r] + b;
      s_h1[mm * CH + col] = (_Float16)(v > 0.f ? v : 0.f);
    }
  }
  __syncthreads();

  // layer 2: [16,128] x [128,64], waves 0..3
  if (w < 4) {
    v8f acc = {};
#pragma unroll
    for (int kt = 0; kt < 4; ++kt) {
      v16h a = load_a_frag(s_h1, m, kt, hh, CH);
      acc = wmma_f16(a, load_b_frag(w2_p, w * 4 + kt, lane), acc);
    }
    int col = w * 16 + m;
    float b = b2[col];
#pragma unroll
    for (int r = 0; r < 8; ++r) {
      int mm = r + hh * 8;
      float v = acc[r] + b;
      s_h2[mm * 64 + col] = v > 0.f ? v : 0.f;
    }
  }
  __syncthreads();

  // layer 3: [16,64] x [64,2] — tiny, vectorized VALU dot per (edge, out)
  if (tid < 32) {
    int er = tid >> 1;
    int o  = tid & 1;
    float sum = b3[o];
#pragma unroll
    for (int kk = 0; kk < 16; ++kk) {
      v4f hvec = *(const v4f*)(s_h2 + er * 64 + kk * 4);
      v4f wv   = *(const v4f*)(s_w3 + o * 64 + kk * 4);
      sum += hvec[0] * wv[0] + hvec[1] * wv[1] + hvec[2] * wv[2] + hvec[3] * wv[3];
    }
    out[(size_t)(e0 + er) * 2 + o] = sum;
  }
}

extern "C" void kernel_launch(void* const* d_in, const int* in_sizes, int n_in,
                              void* d_out, int out_size, void* d_ws, size_t ws_size,
                              hipStream_t stream) {
  const float* x     = (const float*)d_in[0];
  const int*   ei    = (const int*)d_in[1];
  const float* eattr = (const float*)d_in[2];
  const float* Wih   = (const float*)d_in[3];
  const float* Whh   = (const float*)d_in[4];
  const float* bih   = (const float*)d_in[5];
  const float* bhh   = (const float*)d_in[6];
  const float* linlW = (const float*)d_in[7];
  const float* linlb = (const float*)d_in[8];
  const float* linrW = (const float*)d_in[9];
  const float* W1 = (const float*)d_in[10];
  const float* b1 = (const float*)d_in[11];
  const float* W2 = (const float*)d_in[12];
  const float* b2 = (const float*)d_in[13];
  const float* W3 = (const float*)d_in[14];
  const float* b3 = (const float*)d_in[15];
  float* out = (float*)d_out;
  const int* src = ei;
  const int* dst = ei + EDGES;
  (void)in_sizes; (void)n_in; (void)out_size; (void)ws_size;

  char* ws = (char*)d_ws;
  size_t off = 0;
  auto alloc = [&](size_t bytes) -> void* {
    void* p = ws + off;
    off = (off + bytes + 255) & ~(size_t)255;
    return p;
  };
  _Float16* hA     = (_Float16*)alloc((size_t)NODES * CH * sizeof(_Float16));
  _Float16* hB     = (_Float16*)alloc((size_t)NODES * CH * sizeof(_Float16));
  _Float16* wih_p  = (_Float16*)alloc(3 * (size_t)CH * GATES * sizeof(_Float16));
  _Float16* whh_p  = (_Float16*)alloc(3 * (size_t)CH * GATES * sizeof(_Float16));
  _Float16* linl_p = (_Float16*)alloc(3 * (size_t)CH * CH * sizeof(_Float16));
  _Float16* linr_p = (_Float16*)alloc(3 * (size_t)CH * CH * sizeof(_Float16));
  _Float16* w1_p   = (_Float16*)alloc((size_t)K1PAD * CH * sizeof(_Float16));
  _Float16* w2_p   = (_Float16*)alloc((size_t)CH * 64 * sizeof(_Float16));

  // pack all weights to f16 B-fragment layout (L2-resident afterwards)
  const int nIh = CH * GATES;   // 65536 halves per matrix
  const int nLn = CH * CH;      // 16384 halves
  for (int l = 0; l < 3; ++l) {
    pack_b_frag<<<(nIh + 255) / 256, 256, 0, stream>>>(
        Wih + (size_t)l * GATES * CH, wih_p + (size_t)l * nIh, CH, CH, GATES);
    pack_b_frag<<<(nIh + 255) / 256, 256, 0, stream>>>(
        Whh + (size_t)l * GATES * CH, whh_p + (size_t)l * nIh, CH, CH, GATES);
    pack_b_frag<<<(nLn + 255) / 256, 256, 0, stream>>>(
        linlW + (size_t)l * nLn, linl_p + (size_t)l * nLn, CH, CH, CH);
    pack_b_frag<<<(nLn + 255) / 256, 256, 0, stream>>>(
        linrW + (size_t)l * nLn, linr_p + (size_t)l * nLn, CH, CH, CH);
  }
  pack_b_frag<<<((K1PAD * CH) + 255) / 256, 256, 0, stream>>>(W1, w1_p, K1REAL, K1PAD, CH);
  pack_b_frag<<<((CH * 64) + 255) / 256, 256, 0, stream>>>(W2, w2_p, CH, CH, 64);

  dim3 blk(256);
  sage_lstm_layer<<<NODES / 16, blk, 0, stream>>>(
      x, nullptr, 1, src, wih_p, whh_p, bih, bhh, linl_p, linlb, linr_p, hA);
  sage_lstm_layer<<<NODES / 16, blk, 0, stream>>>(
      nullptr, hA, 0, src, wih_p + nIh, whh_p + nIh, bih + GATES, bhh + GATES,
      linl_p + nLn, linlb + CH, linr_p + nLn, hB);
  sage_lstm_layer<<<NODES / 16, blk, 0, stream>>>(
      nullptr, hB, 0, src, wih_p + 2 * nIh, whh_p + 2 * nIh, bih + 2 * GATES,
      bhh + 2 * GATES, linl_p + 2 * nLn, linlb + 2 * CH, linr_p + 2 * nLn, hA);

  edge_mlp<<<EDGES / 16, blk, 0, stream>>>(hA, src, dst, eattr, w1_p, b1,
                                           w2_p, b2, W3, b3, out);
}
